// MultiHeadAttention_9053791060332
// MI455X (gfx1250) — matrix operand order repaired and checked by reading, not compiled, not run on hardware
//
#include <hip/hip_runtime.h>

// Multi-head attention forward for gfx1250 (MI455X).
// bf16 WMMA for all GEMMs + attention, TDM (tensor_load_to_lds) staging of
// shared weight tiles into LDS, flash-style online-softmax attention.
// B=2, T=2048, C=1024, H=16, Dh=64.

#define NEMBD 1024
#define NHEAD 16
#define HDIM  64
#define TSEQ  2048
#define BATCH 2
#define MROWS (BATCH * TSEQ)   // 4096

typedef __bf16 bf16;
typedef __bf16 v16bf __attribute__((ext_vector_type(16)));
typedef __bf16 v8bf  __attribute__((ext_vector_type(8)));
typedef float  v8f   __attribute__((ext_vector_type(8)));
typedef float  v4f   __attribute__((ext_vector_type(4)));
typedef unsigned short v8u  __attribute__((ext_vector_type(8)));
typedef unsigned int   u32x4 __attribute__((ext_vector_type(4)));
typedef int            i32x4 __attribute__((ext_vector_type(4)));
typedef int            i32x8 __attribute__((ext_vector_type(8)));

#if defined(__has_builtin)
#if __has_builtin(__builtin_amdgcn_tensor_load_to_lds) && \
    __has_builtin(__builtin_amdgcn_s_wait_tensorcnt)
#define USE_TDM 1
#endif
#endif
#ifndef USE_TDM
#define USE_TDM 0
#endif

static __device__ __forceinline__ unsigned short f2bf(float f) {
    unsigned int u = __float_as_uint(f);
    u = (u + 0x7FFFu + ((u >> 16) & 1u)) >> 16;   // round-to-nearest-even
    return (unsigned short)u;
}

static __device__ __forceinline__ v8f wmma_bf16(v16bf a, v16bf b, v8f c) {
    // D = A(16x32 bf16) x B(32x16 bf16) + C(16x16 f32)
    return __builtin_amdgcn_wmma_f32_16x16x32_bf16(
        /*neg_a=*/false, a, /*neg_b=*/false, b,
        /*c_mod=*/(short)0, c, /*reuse_a=*/false, /*reuse_b=*/false);
}

static __device__ __forceinline__ v16bf join8(v8bf lo, v8bf hi) {
    return __builtin_shufflevector(lo, hi, 0,1,2,3,4,5,6,7,8,9,10,11,12,13,14,15);
}

#if USE_TDM
// TDM descriptor (D#) for a 2-D tile: 64 rows x 32 cols of bf16, row-major
// source matrix of width 1024 elements. Groups per CDNA5 ISA §8.3/8.4.
static __device__ __forceinline__ void tdm_load_tile_64x32(const unsigned short* gsrc,
                                                           unsigned lds_byte_addr) {
    unsigned long long ga = (unsigned long long)(size_t)gsrc;
    u32x4 g0;
    g0[0] = 1u;                                      // count=1 (valid user D#)
    g0[1] = lds_byte_addr;                           // lds_addr (bytes)
    g0[2] = (unsigned)ga;                            // global_addr[31:0]
    g0[3] = (unsigned)((ga >> 32) & 0x01FFFFFFull)   // global_addr[56:32]
            | (2u << 30);                            // type=2 ("image")
    i32x8 g1;
    g1[0] = 1 << 16;                 // workgroup_mask=0, data_size=1 (2 bytes)
    g1[1] = (int)(0x0400u << 16);    // tensor_dim0[15:0]=1024 in bits 63:48
    g1[2] = (int)(0x0400u << 16);    // dim0[31:16]=0 | tensor_dim1[15:0]=1024
    g1[3] = (int)(32u << 16);        // dim1[31:16]=0 | tile_dim0=32
    g1[4] = 64;                      // tile_dim1=64 | tile_dim2=0
    g1[5] = 1024;                    // tensor_dim0_stride[31:0]=1024
    g1[6] = 0;                       // stride0 hi | tensor_dim1_stride lo
    g1[7] = 0;                       // tensor_dim1_stride hi
    i32x4 zz = {0, 0, 0, 0};         // groups 2/3 unused (2-D tile)
#if __clang_major__ >= 23
    i32x8 z8 = {};
    __builtin_amdgcn_tensor_load_to_lds(g0, g1, zz, zz, z8, 0);
#else
    __builtin_amdgcn_tensor_load_to_lds(g0, g1, zz, zz, 0);
#endif
}
#endif

// ---------------------------------------------------------------------------
// 1) LayerNorm (f32 in) -> bf16 normalized activations, one block per row.
// ---------------------------------------------------------------------------
__global__ void ln_bf16_kernel(const float* __restrict__ x,
                               const float* __restrict__ gamma,
                               const float* __restrict__ beta,
                               unsigned short* __restrict__ xn) {
    __shared__ float r1[256], r2[256];
    const int row = blockIdx.x;
    const int t   = threadIdx.x;
    const float* xr = x + (size_t)row * NEMBD;
    v4f v = *(const v4f*)(xr + t * 4);
    float s1 = v[0] + v[1] + v[2] + v[3];
    float s2 = v[0]*v[0] + v[1]*v[1] + v[2]*v[2] + v[3]*v[3];
    r1[t] = s1; r2[t] = s2;
    __syncthreads();
    for (int off = 128; off > 0; off >>= 1) {
        if (t < off) { r1[t] += r1[t + off]; r2[t] += r2[t + off]; }
        __syncthreads();
    }
    const float mean = r1[0] * (1.0f / NEMBD);
    const float var  = r2[0] * (1.0f / NEMBD) - mean * mean;
    const float rstd = rsqrtf(var + 1e-5f);
#pragma unroll
    for (int j = 0; j < 4; ++j) {
        const int c = t * 4 + j;
        xn[(size_t)row * NEMBD + c] = f2bf((v[j] - mean) * rstd * gamma[c] + beta[c]);
    }
}

// ---------------------------------------------------------------------------
// 2) f32 -> bf16 weight conversion (1M elements per matrix; 4 per thread).
// ---------------------------------------------------------------------------
__global__ void cvt_bf16_kernel(const float* __restrict__ src,
                                unsigned short* __restrict__ dst) {
    const int i = (blockIdx.x * blockDim.x + threadIdx.x) * 4;
    v4f v = *(const v4f*)(src + i);
#pragma unroll
    for (int j = 0; j < 4; ++j) dst[i + j] = f2bf(v[j]);
}

// ---------------------------------------------------------------------------
// 3) Fused QKV NT-GEMM: y = xn @ W^T. Block = 8 waves, tile 128(M) x 64(N).
//    The shared 64x32 W tile per K-step is staged through LDS by the TDM
//    (double-buffered; one tensor_load_to_lds per step issued by wave 0,
//    s_wait_tensorcnt + barrier publishes it, DMA of step k+1 overlaps the
//    4 WMMAs of step k). A-fragments are per-wave-unique -> direct global.
//    Q,K stored [B,H,T,64] bf16; V stored transposed [B,H,64,T] bf16.
// ---------------------------------------------------------------------------
__global__ void qkv_kernel(const unsigned short* __restrict__ xn,
                           const unsigned short* __restrict__ wq,
                           const unsigned short* __restrict__ wk,
                           const unsigned short* __restrict__ wv,
                           unsigned short* __restrict__ Q,
                           unsigned short* __restrict__ K,
                           unsigned short* __restrict__ Vt) {
#if USE_TDM
    __shared__ __attribute__((aligned(64))) unsigned short wtile[2][64][32];
#endif
    const int wave = threadIdx.x >> 5;
    const int lane = threadIdx.x & 31;
    const int half = lane >> 4;
    const int l16  = lane & 15;
    const int m0 = blockIdx.x * 128 + wave * 16;
    const int n0 = blockIdx.y * 64;
    const int sel = blockIdx.z;
    const unsigned short* W = (sel == 0) ? wq : ((sel == 1) ? wk : wv);

    v8f acc[4] = {};
    // A fragment: lane = row (l16), half selects 16 contiguous K {h*16..h*16+15}
    const unsigned short* arow = xn + (size_t)(m0 + l16) * NEMBD + half * 16;

#if USE_TDM
    const unsigned short* wt = W + (size_t)n0 * NEMBD;
    if (wave == 0)
        tdm_load_tile_64x32(wt, (unsigned)(size_t)(void*)&wtile[0][0][0]);
    int cur = 0;
    for (int k = 0; k < NEMBD; k += 32) {
        if (wave == 0) __builtin_amdgcn_s_wait_tensorcnt(0);
        __syncthreads();
        if (wave == 0 && (k + 32) < NEMBD)
            tdm_load_tile_64x32(wt + (k + 32),
                                (unsigned)(size_t)(void*)&wtile[cur ^ 1][0][0]);
        __builtin_prefetch(arow + k + 128, 0, 0);
        v8bf alo = *(const v8bf*)(arow + k);
        v8bf ahi = *(const v8bf*)(arow + k + 8);
        v16bf af = join8(alo, ahi);
#pragma unroll
        for (int a = 0; a < 4; ++a) {
            // B fragment from LDS: lane = W row (a*16+l16), 16 contiguous K
            v16bf bfr = *(const v16bf*)&wtile[cur][a * 16 + l16][half * 16];
            acc[a] = wmma_bf16(af, bfr, acc[a]);
        }
        cur ^= 1;
    }
#else
    for (int k = 0; k < NEMBD; k += 32) {
        __builtin_prefetch(arow + k + 128, 0, 0);
        v8bf alo = *(const v8bf*)(arow + k);
        v8bf ahi = *(const v8bf*)(arow + k + 8);
        v16bf af = join8(alo, ahi);
#pragma unroll
        for (int a = 0; a < 4; ++a) {
            const unsigned short* brow =
                W + (size_t)(n0 + a * 16 + l16) * NEMBD + k + half * 16;
            acc[a] = wmma_bf16(af, *(const v16bf*)brow, acc[a]);
        }
    }
#endif

    if (sel < 2) {
        unsigned short* dst = (sel == 0) ? Q : K;
#pragma unroll
        for (int a = 0; a < 4; ++a) {
            const int gn = n0 + a * 16 + l16;
            const int h = gn >> 6, dh = gn & 63;
#pragma unroll
            for (int r = 0; r < 8; ++r) {
                const int gm = m0 + half * 8 + r;
                const int b = gm >> 11, tt = gm & (TSEQ - 1);
                dst[(((size_t)(b * NHEAD + h) * TSEQ + tt) << 6) + dh] = f2bf(acc[a][r]);
            }
        }
    } else {
        // V transposed: lane holds fixed (h,dh); 8 consecutive t -> 16B store
#pragma unroll
        for (int a = 0; a < 4; ++a) {
            const int gn = n0 + a * 16 + l16;
            const int h = gn >> 6, dh = gn & 63;
            const int gm0 = m0 + half * 8;
            const int b = gm0 >> 11, t0 = gm0 & (TSEQ - 1);
            v8u pack;
#pragma unroll
            for (int r = 0; r < 8; ++r) pack[r] = f2bf(acc[a][r]);
            *(v8u*)(Vt + ((size_t)(b * NHEAD + h) * HDIM + dh) * TSEQ + t0) = pack;
        }
    }
}

// ---------------------------------------------------------------------------
// 4) Flash-style causal attention. One wave = 16 query rows; 32-key blocks:
//    4 wmmas for S = Q K^T, online softmax (cross-lane reductions within
//    16-lane halves match the C/D layout), P transposed via per-wave LDS tile,
//    4 wmmas for O += P V. Output bf16 [B,T,C].
// ---------------------------------------------------------------------------
__global__ void attn_kernel(const unsigned short* __restrict__ Q,
                            const unsigned short* __restrict__ K,
                            const unsigned short* __restrict__ Vt,
                            unsigned short* __restrict__ AO) {
    __shared__ __attribute__((aligned(64))) unsigned short lds_p[8][16][32];
    const int wave = threadIdx.x >> 5;
    const int lane = threadIdx.x & 31;
    const int half = lane >> 4;
    const int l16  = lane & 15;

    const int tiles = TSEQ / 128;                 // 16
    const int bh  = blockIdx.x / tiles;           // b*H + h
    const int tq0 = (blockIdx.x % tiles) * 128 + wave * 16;

    const unsigned short* Qh = Q  + ((size_t)bh * TSEQ << 6);
    const unsigned short* Kh = K  + ((size_t)bh * TSEQ << 6);
    const unsigned short* Vh = Vt + ((size_t)bh << 6) * TSEQ;

    // Q A-fragments for the two Dh K-steps
    v16bf qf[2];
    {
        const unsigned short* qrow = Qh + ((size_t)(tq0 + l16) << 6) + half * 16;
#pragma unroll
        for (int kk = 0; kk < 2; ++kk) {
            v8bf lo = *(const v8bf*)(qrow + kk * 32);
            v8bf hi = *(const v8bf*)(qrow + kk * 32 + 8);
            qf[kk] = join8(lo, hi);
        }
    }

    v8f o[4] = {};
    float mrow[8], lrow[8];
#pragma unroll
    for (int r = 0; r < 8; ++r) { mrow[r] = -1e30f; lrow[r] = 0.0f; }

    const int nkb = (tq0 + 15) / 32 + 1;          // causal: keys <= tq0+15
    for (int kb = 0; kb < nkb; ++kb) {
        const int j0 = kb * 32;
        // ---- scores: two 16-col subtiles, K-dim = Dh = 64 (2 wmma steps) ----
        v8f s[2] = {};
#pragma unroll
        for (int ss = 0; ss < 2; ++ss) {
            const unsigned short* krow = Kh + ((size_t)(j0 + ss * 16 + l16) << 6);
#pragma unroll
            for (int kk = 0; kk < 2; ++kk) {
                v16bf bfr = *(const v16bf*)(krow + kk * 32 + half * 16);
                s[ss] = wmma_bf16(qf[kk], bfr, s[ss]);
            }
        }
        // ---- scale + causal mask + per-row partial max ----
        float pm[8];
#pragma unroll
        for (int r = 0; r < 8; ++r) {
            const int qt = tq0 + half * 8 + r;
#pragma unroll
            for (int ss = 0; ss < 2; ++ss) {
                const int kt = j0 + ss * 16 + l16;
                float v = s[ss][r] * 0.125f;      // 1/sqrt(64)
                s[ss][r] = (kt <= qt) ? v : -1e30f;
            }
            pm[r] = fmaxf(s[0][r], s[1][r]);
        }
#pragma unroll
        for (int d = 1; d < 16; d <<= 1) {
#pragma unroll
            for (int r = 0; r < 8; ++r)
                pm[r] = fmaxf(pm[r], __shfl_xor(pm[r], d, 32));
        }
        // ---- online softmax update ----
        float alpha[8], ps[8];
#pragma unroll
        for (int r = 0; r < 8; ++r) {
            const float mnew = fmaxf(mrow[r], pm[r]);
            alpha[r] = __expf(mrow[r] - mnew);
            mrow[r] = mnew;
            const float p0 = __expf(s[0][r] - mnew);
            const float p1 = __expf(s[1][r] - mnew);
            s[0][r] = p0; s[1][r] = p1;
            ps[r] = p0 + p1;
        }
#pragma unroll
        for (int d = 1; d < 16; d <<= 1) {
#pragma unroll
            for (int r = 0; r < 8; ++r)
                ps[r] += __shfl_xor(ps[r], d, 32);
        }
#pragma unroll
        for (int r = 0; r < 8; ++r) lrow[r] = lrow[r] * alpha[r] + ps[r];
#pragma unroll
        for (int a = 0; a < 4; ++a) {
#pragma unroll
            for (int r = 0; r < 8; ++r) o[a][r] *= alpha[r];
        }
        // ---- P: C-layout -> A-layout transpose through per-wave LDS tile ----
#pragma unroll
        for (int ss = 0; ss < 2; ++ss) {
#pragma unroll
            for (int r = 0; r < 8; ++r)
                lds_p[wave][half * 8 + r][ss * 16 + l16] = f2bf(s[ss][r]);
        }
        // DS ops are in-order per wave; dependence through lds_p keeps order.
        v8bf plo = *(const v8bf*)&lds_p[wave][l16][half * 16];
        v8bf phi = *(const v8bf*)&lds_p[wave][l16][half * 16 + 8];
        v16bf pf = join8(plo, phi);
        // ---- O += P (16x32) x V (32x64): 4 wmmas, V^T rows contiguous ----
#pragma unroll
        for (int a = 0; a < 4; ++a) {
            const unsigned short* vrow =
                Vh + (size_t)(a * 16 + l16) * TSEQ + j0 + half * 16;
            o[a] = wmma_bf16(pf, *(const v16bf*)vrow, o[a]);
        }
    }

    // ---- normalize and store as [B,T,C] bf16 for the output projection ----
#pragma unroll
    for (int r = 0; r < 8; ++r) lrow[r] = 1.0f / lrow[r];
    const int b = bh / NHEAD, h = bh % NHEAD;
#pragma unroll
    for (int a = 0; a < 4; ++a) {
        const int c = h * 64 + a * 16 + l16;
#pragma unroll
        for (int r = 0; r < 8; ++r) {
            const int t = tq0 + half * 8 + r;
            AO[(size_t)(b * TSEQ + t) * NEMBD + c] = f2bf(o[a][r] * lrow[r]);
        }
    }
}

// ---------------------------------------------------------------------------
// 5) Output projection NT-GEMM: out = AO @ Wp^T + bp (f32 output).
//    Same TDM-staged weight tiles as the QKV GEMM.
// ---------------------------------------------------------------------------
__global__ void proj_kernel(const unsigned short* __restrict__ A,
                            const unsigned short* __restrict__ W,
                            const float* __restrict__ bias,
                            float* __restrict__ out) {
#if USE_TDM
    __shared__ __attribute__((aligned(64))) unsigned short wtile[2][64][32];
#endif
    const int wave = threadIdx.x >> 5;
    const int lane = threadIdx.x & 31;
    const int half = lane >> 4;
    const int l16  = lane & 15;
    const int m0 = blockIdx.x * 128 + wave * 16;
    const int n0 = blockIdx.y * 64;

    v8f acc[4] = {};
    const unsigned short* arow = A + (size_t)(m0 + l16) * NEMBD + half * 16;

#if USE_TDM
    const unsigned short* wt = W + (size_t)n0 * NEMBD;
    if (wave == 0)
        tdm_load_tile_64x32(wt, (unsigned)(size_t)(void*)&wtile[0][0][0]);
    int cur = 0;
    for (int k = 0; k < NEMBD; k += 32) {
        if (wave == 0) __builtin_amdgcn_s_wait_tensorcnt(0);
        __syncthreads();
        if (wave == 0 && (k + 32) < NEMBD)
            tdm_load_tile_64x32(wt + (k + 32),
                                (unsigned)(size_t)(void*)&wtile[cur ^ 1][0][0]);
        __builtin_prefetch(arow + k + 128, 0, 0);
        v8bf alo = *(const v8bf*)(arow + k);
        v8bf ahi = *(const v8bf*)(arow + k + 8);
        v16bf af = join8(alo, ahi);
#pragma unroll
        for (int a = 0; a < 4; ++a) {
            v16bf bfr = *(const v16bf*)&wtile[cur][a * 16 + l16][half * 16];
            acc[a] = wmma_bf16(af, bfr, acc[a]);
        }
        cur ^= 1;
    }
#else
    for (int k = 0; k < NEMBD; k += 32) {
        __builtin_prefetch(arow + k + 128, 0, 0);
        v8bf alo = *(const v8bf*)(arow + k);
        v8bf ahi = *(const v8bf*)(arow + k + 8);
        v16bf af = join8(alo, ahi);
#pragma unroll
        for (int a = 0; a < 4; ++a) {
            const unsigned short* brow =
                W + (size_t)(n0 + a * 16 + l16) * NEMBD + k + half * 16;
            acc[a] = wmma_bf16(af, *(const v16bf*)brow, acc[a]);
        }
    }
#endif

#pragma unroll
    for (int a = 0; a < 4; ++a) {
        const int gn = n0 + a * 16 + l16;
        const float bb = bias[gn];
#pragma unroll
        for (int r = 0; r < 8; ++r) {
            const int gm = m0 + half * 8 + r;
            out[(size_t)gm * NEMBD + gn] = acc[a][r] + bb;
        }
    }
}

// ---------------------------------------------------------------------------
extern "C" void kernel_launch(void* const* d_in, const int* in_sizes, int n_in,
                              void* d_out, int out_size, void* d_ws, size_t ws_size,
                              hipStream_t stream) {
    (void)in_sizes; (void)n_in; (void)out_size; (void)ws_size;
    const float* x     = (const float*)d_in[0];
    const float* Wq    = (const float*)d_in[1];
    const float* Wk    = (const float*)d_in[2];
    const float* Wv    = (const float*)d_in[3];
    const float* Wp    = (const float*)d_in[4];
    const float* bp    = (const float*)d_in[5];
    const float* gamma = (const float*)d_in[6];
    const float* beta  = (const float*)d_in[7];
    float* out = (float*)d_out;

    char* ws = (char*)d_ws;
    const size_t MB = 1u << 20;
    unsigned short* xn  = (unsigned short*)(ws + 0);        //  8 MiB
    unsigned short* wqb = (unsigned short*)(ws +  8 * MB);  //  2 MiB
    unsigned short* wkb = (unsigned short*)(ws + 10 * MB);
    unsigned short* wvb = (unsigned short*)(ws + 12 * MB);
    unsigned short* wpb = (unsigned short*)(ws + 14 * MB);
    unsigned short* Qb  = (unsigned short*)(ws + 16 * MB);  //  8 MiB
    unsigned short* Kb  = (unsigned short*)(ws + 24 * MB);  //  8 MiB
    unsigned short* Vtb = (unsigned short*)(ws + 32 * MB);  //  8 MiB
    unsigned short* AO  = xn;   // alias: xn fully consumed before attn writes

    ln_bf16_kernel<<<MROWS, 256, 0, stream>>>(x, gamma, beta, xn);
    cvt_bf16_kernel<<<1024, 256, 0, stream>>>(Wq, wqb);
    cvt_bf16_kernel<<<1024, 256, 0, stream>>>(Wk, wkb);
    cvt_bf16_kernel<<<1024, 256, 0, stream>>>(Wv, wvb);
    cvt_bf16_kernel<<<1024, 256, 0, stream>>>(Wp, wpb);

    dim3 gqkv(MROWS / 128, NEMBD / 64, 3);
    qkv_kernel<<<gqkv, 256, 0, stream>>>(xn, wqb, wkb, wvb, Qb, Kb, Vtb);

    attn_kernel<<<BATCH * NHEAD * (TSEQ / 128), 256, 0, stream>>>(Qb, Kb, Vtb, AO);

    dim3 gproj(MROWS / 128, NEMBD / 64, 1);
    proj_kernel<<<gproj, 256, 0, stream>>>(AO, wpb, bp, out);
}
